// SimpleRNN_89000312307751
// MI455X (gfx1250) — matrix operand order repaired and checked by reading, not compiled, not run on hardware
//
#include <hip/hip_runtime.h>
#include <cstdint>
#include <cstddef>

// ---------------- problem dims ----------------
#define BB   16
#define TT   256
#define HH   256
#define H3   768
#define DE   100
#define DFF  1024
#define VOUT 32000
#define NROWS (BB*TT)          // 4096 token rows

typedef __bf16 bf16_t;
typedef __bf16        v16bf __attribute__((ext_vector_type(16)));
typedef float         v8f   __attribute__((ext_vector_type(8)));
typedef unsigned int  u32x4 __attribute__((ext_vector_type(4)));
typedef unsigned int  u32x2 __attribute__((ext_vector_type(2)));
typedef int           i4v   __attribute__((vector_size(16)));   // matches builtin sig

#if __has_builtin(__builtin_amdgcn_global_load_async_to_lds_b128)
#define HAVE_ASYNC_LDS 1
#else
#define HAVE_ASYNC_LDS 0
#endif

static __device__ __forceinline__ void wait_asynccnt0() {
#if __has_builtin(__builtin_amdgcn_s_wait_asynccnt)
    __builtin_amdgcn_s_wait_asynccnt(0);
#else
    asm volatile("s_wait_asynccnt 0" ::: "memory");
#endif
}

static __device__ __forceinline__ v16bf mk16(u32x4 a, u32x4 b) {
    union { u32x4 q[2]; v16bf v; } u; u.q[0] = a; u.q[1] = b; return u.v;
}
static __device__ __forceinline__ v8f wmma_bf16(v16bf a, v16bf b, v8f c) {
    // D = A(16x32 bf16) * B(32x16 bf16) + C(f32)
    return __builtin_amdgcn_wmma_f32_16x16x32_bf16(false, a, false, b, (short)0, c, false, false);
}

// A-operand (16x32 bf16) from row-major bf16 buffer (global or LDS).
// lane L: row m=L%16; two 16B chunks at k = kbase + (L/16)*8 and +16.
static __device__ __forceinline__ v16bf loadA(const bf16_t* base, int pitch,
                                              int row0, int kbase, int lane) {
    int m = lane & 15, g = lane >> 4;
    const bf16_t* p = base + (size_t)(row0 + m) * pitch + kbase + g * 8;
    u32x4 lo = *(const u32x4*)p;
    u32x4 hi = *(const u32x4*)(p + 16);
    return mk16(lo, hi);
}

// B-operand from pre-swizzled tile blocks: block = 32 lanes x 16 bf16 (1KB),
// lane's 16 values contiguous (k = kt*32 + (v/4)*16 + (L/16)*8 + 2(v%4), n = nt*16 + L%16).
static __device__ __forceinline__ v16bf loadB(const bf16_t* tiled, int block, int lane) {
    const bf16_t* p = tiled + ((size_t)block * 32 + lane) * 16;
    u32x4 lo = *(const u32x4*)p;
    u32x4 hi = *(const u32x4*)(p + 8);
    return mk16(lo, hi);
}

// ---------------- K0: fp32 weights -> bf16 WMMA B-tile layout ----------------
__global__ void convert_tileB(const float* __restrict__ src, bf16_t* __restrict__ dst,
                              int K, int N) {
    long long p = (long long)blockIdx.x * 256 + threadIdx.x;   // pair index
    long long npairs = (long long)K * N / 2;
    if (p >= npairs) return;
    int v = (int)(p & 7);
    int L = (int)((p >> 3) & 31);
    long long q = p >> 8;
    int KT = K >> 5;
    int kt = (int)(q % KT);
    long long nt = q / KT;
    int n = (int)nt * 16 + (L & 15);
    int k = kt * 32 + (v >> 2) * 16 + (L >> 4) * 8 + 2 * (v & 3);
    union { bf16_t b[2]; unsigned int u; } pk;
    pk.b[0] = (bf16_t)src[(size_t)k * N + n];
    pk.b[1] = (bf16_t)src[(size_t)(k + 1) * N + n];
    ((unsigned int*)dst)[(((nt * KT + kt) * 32 + L) * 16 + 2 * v) >> 1] = pk.u;
}

// ---------------- K1: embedding gather + x @ Wx + b0 -> xproj f32 ------------
__global__ void embed_xproj(const int* __restrict__ tokens, const float* __restrict__ emb,
                            const float* __restrict__ Wx, const float* __restrict__ b0,
                            float* __restrict__ xproj) {
    __shared__ float se[DE];
    int row = blockIdx.x;
    int tok = tokens[row];
    for (int d = threadIdx.x; d < DE; d += 256) se[d] = emb[(size_t)tok * DE + d];
    __syncthreads();
    for (int j = threadIdx.x; j < H3; j += 256) {
        float acc = b0[j];
        #pragma unroll 4
        for (int d = 0; d < DE; d++) acc = fmaf(se[d], Wx[(size_t)d * H3 + j], acc);
        xproj[(size_t)row * H3 + j] = acc;
    }
}

// ---------------- K2: GRU scan (single persistent WG, WMMA recurrence) -------
// 16 waves; wave w computes rec N-tiles 3w..3w+2 ([16,256]x[256,768] per step).
// LLVM hoists the 24 B-tiles (tiled Wh) into ~192 VGPRs: register-resident
// recurrent weights; per step only A comes from LDS (ds_load_b128 x2 / k-step).
__global__ void __launch_bounds__(512, 1)
gru_scan(const float* __restrict__ xproj, const bf16_t* __restrict__ whT,
         const float* __restrict__ brec, bf16_t* __restrict__ hseq) {
    __shared__ float  s_rec[BB * H3];      // 48 KB, rec = h @ Wh
    __shared__ bf16_t s_h[BB * 264];       // h state, padded pitch (bank spread)
    const int tid = threadIdx.x, w = tid >> 5, lane = tid & 31;

    for (int i = tid; i < BB * 264; i += 512) s_h[i] = (bf16_t)0.0f;
    __syncthreads();

    for (int t = 0; t < TT; t++) {
        v8f c0 = {}, c1 = {}, c2 = {};
        const int nt0 = w * 3;
        #pragma unroll
        for (int kt = 0; kt < 8; kt++) {
            v16bf a  = loadA(s_h, 264, 0, kt * 32, lane);      // ds_load_b128 x2
            v16bf b0t = loadB(whT, (nt0 + 0) * 8 + kt, lane);
            c0 = wmma_bf16(a, b0t, c0);
            v16bf b1t = loadB(whT, (nt0 + 1) * 8 + kt, lane);
            c1 = wmma_bf16(a, b1t, c1);
            v16bf b2t = loadB(whT, (nt0 + 2) * 8 + kt, lane);
            c2 = wmma_bf16(a, b2t, c2);
        }
        {   // C layout: VGPR v -> row v + 8*(lane/16), col lane%16
            int n = lane & 15, g = lane >> 4;
            #pragma unroll
            for (int v = 0; v < 8; v++) {
                int m = v + g * 8;
                s_rec[m * H3 + (nt0 + 0) * 16 + n] = c0[v];
                s_rec[m * H3 + (nt0 + 1) * 16 + n] = c1[v];
                s_rec[m * H3 + (nt0 + 2) * 16 + n] = c2[v];
            }
        }
        __syncthreads();
        // gates: 4096 elements over 512 threads (uniform 8 iterations)
        for (int e = tid; e < BB * HH; e += 512) {
            int b = e >> 8, i = e & 255;
            const float* xp = xproj + (size_t)(b * TT + t) * H3;
            float rz = s_rec[b * H3 + i]        + brec[i];
            float rr = s_rec[b * H3 + 256 + i]  + brec[256 + i];
            float rh = s_rec[b * H3 + 512 + i]  + brec[512 + i];
            float z  = 1.f / (1.f + __expf(-(xp[i] + rz)));
            float r  = 1.f / (1.f + __expf(-(xp[256 + i] + rr)));
            float pre = xp[512 + i] + r * rh;
            pre = fminf(fmaxf(pre, -15.f), 15.f);
            float e2 = __expf(2.f * pre);
            float hh = (e2 - 1.f) / (e2 + 1.f);
            float hp = (float)s_h[b * 264 + i];
            float hn = z * hp + (1.f - z) * hh;
            s_h[b * 264 + i] = (bf16_t)hn;
            hseq[(size_t)(b * TT + t) * HH + i] = (bf16_t)hn;
        }
        __syncthreads();
    }
}

// ---------------- K3: d1 = relu(h_seq @ W1 + b1), bf16 out -------------------
__global__ void ffn1(const bf16_t* __restrict__ hseq, const bf16_t* __restrict__ w1T,
                     const float* __restrict__ b1, bf16_t* __restrict__ d1) {
    int gw = blockIdx.x * 8 + (threadIdx.x >> 5), lane = threadIdx.x & 31;
    int mt = gw >> 6, nt = gw & 63;           // 256 x 64 tiles
    v8f c = {};
    #pragma unroll
    for (int kt = 0; kt < 8; kt++) {
        v16bf a = loadA(hseq, HH, mt * 16, kt * 32, lane);
        v16bf b = loadB(w1T, nt * 8 + kt, lane);
        c = wmma_bf16(a, b, c);
    }
    int n = lane & 15, g = lane >> 4;
    #pragma unroll
    for (int v = 0; v < 8; v++) {
        int row = mt * 16 + v + g * 8;
        int col = nt * 16 + n;
        float x = c[v] + b1[col];
        x = x > 0.f ? x : 0.f;
        d1[(size_t)row * DFF + col] = (bf16_t)x;
    }
}

// ---------------- K4a: logits = d1 @ W2 + b2 (fp32 to d_out) -----------------
// Block = 64 rows x 6400 cols (25 iters of 256 cols). A chunk staged in LDS
// via CDNA5 async global->LDS copies (ASYNCcnt) when the builtin is available.
__global__ void __launch_bounds__(256, 1)
ffn2_logits(const bf16_t* __restrict__ d1, const bf16_t* __restrict__ w2T,
            const float* __restrict__ b2, float* __restrict__ out) {
    __shared__ bf16_t sA[64 * 264];                       // 33 KB, padded pitch
    const int tid = threadIdx.x, w = tid >> 5, lane = tid & 31;
    const int mtg = w >> 2, ntg = w & 3;                  // 2 x 4 wave grid
    const int rowbase = blockIdx.x * 64;
    const int colsplit = blockIdx.y * 6400;

    for (int it = 0; it < 25; it++) {
        const int colbase = colsplit + it * 256;
        v8f c[2][4] = {};
        for (int kc = 0; kc < 4; kc++) {                  // K chunks of 256
            __syncthreads();
#if HAVE_ASYNC_LDS
            // 64x256 bf16 = 32 KB; 2048 x 16B async descriptors, 8 per thread
            #pragma unroll
            for (int i = 0; i < 8; i++) {
                int j = tid + i * 256;
                int row = j >> 5, c8 = j & 31;
                __attribute__((address_space(1))) i4v* gsrc =
                    (__attribute__((address_space(1))) i4v*)
                    &d1[(size_t)(rowbase + row) * DFF + kc * 256 + c8 * 8];
                __attribute__((address_space(3))) i4v* ldst =
                    (__attribute__((address_space(3))) i4v*)&sA[row * 264 + c8 * 8];
                __builtin_amdgcn_global_load_async_to_lds_b128(gsrc, ldst, 0, 0);
            }
            wait_asynccnt0();
#else
            for (int j = tid; j < 4096; j += 256) {       // fallback: 64x64 u32x2
                int row = j >> 6, c4 = j & 63;
                *(u32x2*)&sA[row * 264 + c4 * 4] =
                    *(const u32x2*)&d1[(size_t)(rowbase + row) * DFF + kc * 256 + c4 * 4];
            }
#endif
            __syncthreads();
            #pragma unroll
            for (int kt = 0; kt < 8; kt++) {
                v16bf a0 = loadA(sA, 264, mtg * 32 + 0,  kt * 32, lane);
                v16bf a1 = loadA(sA, 264, mtg * 32 + 16, kt * 32, lane);
                int ktg = kc * 8 + kt;
                #pragma unroll
                for (int ntl = 0; ntl < 4; ntl++) {
                    int ntglob = (colbase >> 4) + ntg * 4 + ntl;
                    if (ktg + 1 < 32)                      // prefetch next K of B
                        __builtin_prefetch(w2T + ((size_t)(ntglob * 32 + ktg + 1) * 32 + lane) * 16, 0, 0);
                    v16bf b = loadB(w2T, ntglob * 32 + ktg, lane);
                    c[0][ntl] = wmma_bf16(a0, b, c[0][ntl]);
                    c[1][ntl] = wmma_bf16(a1, b, c[1][ntl]);
                }
            }
        }
        int n = lane & 15, g = lane >> 4;
        #pragma unroll
        for (int mt = 0; mt < 2; mt++)
            #pragma unroll
            for (int ntl = 0; ntl < 4; ntl++)
                #pragma unroll
                for (int v = 0; v < 8; v++) {
                    int row = rowbase + mtg * 32 + mt * 16 + v + g * 8;
                    int col = colbase + ntg * 64 + ntl * 16 + n;
                    out[(size_t)row * VOUT + col] = c[mt][ntl][v] + b2[col];
                }
    }
}

// ---------------- K4c: per-row max and 1/sum(exp(x-max)), b128 streams -------
__global__ void softmax_rowstats(const float* __restrict__ out,
                                 float* __restrict__ rmax, float* __restrict__ rinv) {
    __shared__ float red[256];
    int r = blockIdx.x;
    const float4* p4 = (const float4*)(out + (size_t)r * VOUT);  // 8000 float4/row
    float m = -1e30f;
    for (int j = threadIdx.x; j < VOUT / 4; j += 256) {
        float4 v = p4[j];
        m = fmaxf(m, fmaxf(fmaxf(v.x, v.y), fmaxf(v.z, v.w)));
    }
    red[threadIdx.x] = m; __syncthreads();
    for (int s = 128; s > 0; s >>= 1) {
        if (threadIdx.x < s) red[threadIdx.x] = fmaxf(red[threadIdx.x], red[threadIdx.x + s]);
        __syncthreads();
    }
    float mx = red[0]; __syncthreads();
    float sacc = 0.f;
    for (int j = threadIdx.x; j < VOUT / 4; j += 256) {
        float4 v = p4[j];
        sacc += __expf(v.x - mx) + __expf(v.y - mx) + __expf(v.z - mx) + __expf(v.w - mx);
    }
    red[threadIdx.x] = sacc; __syncthreads();
    for (int s = 128; s > 0; s >>= 1) {
        if (threadIdx.x < s) red[threadIdx.x] += red[threadIdx.x + s];
        __syncthreads();
    }
    if (threadIdx.x == 0) { rmax[r] = mx; rinv[r] = 1.f / red[0]; }
}

// ---------------- K4e: out = exp(x - max) * invsum, b128 streams -------------
__global__ void softmax_norm(float* __restrict__ out, const float* __restrict__ rmax,
                             const float* __restrict__ rinv) {
    size_t total4 = (size_t)NROWS * VOUT / 4;              // 32.77M float4
    float4* o4 = (float4*)out;
    for (size_t i4 = (size_t)blockIdx.x * 256 + threadIdx.x; i4 < total4;
         i4 += (size_t)gridDim.x * 256) {
        int r = (int)(i4 / (VOUT / 4));
        float mx = rmax[r], iv = rinv[r];
        float4 v = o4[i4];
        v.x = __expf(v.x - mx) * iv;
        v.y = __expf(v.y - mx) * iv;
        v.z = __expf(v.z - mx) * iv;
        v.w = __expf(v.w - mx) * iv;
        o4[i4] = v;
    }
}

// ---------------- host-side launcher ----------------------------------------
extern "C" void kernel_launch(void* const* d_in, const int* in_sizes, int n_in,
                              void* d_out, int out_size, void* d_ws, size_t ws_size,
                              hipStream_t stream) {
    const int*   tokens = (const int*)  d_in[0];
    const float* emb    = (const float*)d_in[1];
    const float* Wx     = (const float*)d_in[2];
    const float* Wh     = (const float*)d_in[3];
    const float* bgate  = (const float*)d_in[4];   // [2][768]: b[0]=input, b[1]=recurrent
    const float* W1     = (const float*)d_in[5];
    const float* b1     = (const float*)d_in[6];
    const float* W2     = (const float*)d_in[7];
    const float* b2     = (const float*)d_in[8];
    float* out = (float*)d_out;

    // workspace carve-out (~90 MB total)
    char* ws = (char*)d_ws;
    size_t o = 0;
    auto alloc = [&](size_t bytes) { size_t r = o; o = (o + bytes + 255) & ~(size_t)255; return r; };
    float*  xproj = (float*) (ws + alloc((size_t)NROWS * H3 * 4));        // 12.6 MB
    bf16_t* whT   = (bf16_t*)(ws + alloc((size_t)HH * H3 * 2));           // 384 KB
    bf16_t* w1T   = (bf16_t*)(ws + alloc((size_t)HH * DFF * 2));          // 512 KB
    bf16_t* w2T   = (bf16_t*)(ws + alloc((size_t)DFF * VOUT * 2));        // 65.5 MB
    bf16_t* hseq  = (bf16_t*)(ws + alloc((size_t)NROWS * HH * 2));        // 2 MB
    bf16_t* d1    = (bf16_t*)(ws + alloc((size_t)NROWS * DFF * 2));       // 8.4 MB
    float*  rmax  = (float*) (ws + alloc((size_t)NROWS * 4));
    float*  rinv  = (float*) (ws + alloc((size_t)NROWS * 4));

    // K0: weight conversion to bf16 tiled-B layout (one-time per call, cheap)
    convert_tileB<<<(HH * H3 / 2 + 255) / 256, 256, 0, stream>>>(Wh, whT, HH, H3);
    convert_tileB<<<(HH * DFF / 2 + 255) / 256, 256, 0, stream>>>(W1, w1T, HH, DFF);
    convert_tileB<<<((size_t)DFF * VOUT / 2 + 255) / 256, 256, 0, stream>>>(W2, w2T, DFF, VOUT);

    // K1: embedding + input projection
    embed_xproj<<<NROWS, 256, 0, stream>>>(tokens, emb, Wx, bgate, xproj);

    // K2: sequential GRU (single persistent workgroup, WMMA recurrence)
    gru_scan<<<1, 512, 0, stream>>>(xproj, whT, bgate + H3, hseq);

    // K3: FF1 + ReLU
    ffn1<<<(256 * 64) / 8, 256, 0, stream>>>(hseq, w1T, b1, d1);

    // K4: logits GEMM (dominant: 268 GFLOP, bf16 WMMA, W2 L2-resident) + softmax
    dim3 g4(NROWS / 64, 5);
    ffn2_logits<<<g4, 256, 0, stream>>>(d1, w2T, b2, out);
    softmax_rowstats<<<NROWS, 256, 0, stream>>>(out, rmax, rinv);
    softmax_norm<<<32768, 256, 0, stream>>>(out, rmax, rinv);
}
